// SinkhornKnoppProjection_43868795961516
// MI455X (gfx1250) — matrix operand order repaired and checked by reading, not compiled, not run on hardware
//
#include <hip/hip_runtime.h>

// ---------------------------------------------------------------------------
// Sinkhorn-Knopp, B=128, N=M=512, 20 iterations.
// x_t = diag(r) * K * diag(c),  K = exp(z - rowmax)  (kept in LDS as bf16).
// Per batch: 8 workgroups, each owns a 64-row slice of K (row-major) and the
// matching 512x64 slice of K^T, both in LDS. Iterations are WMMA matvecs
// (v_wmma_f32_16x16x32_bf16) with the vector broadcast into all 16 B columns.
// Cross-slice col-sum coupling via L2-resident partials + 8-WG spin barrier.
// HBM traffic: read z once (stays L2-resident for the exact-exp final pass,
// which uses NT loads/stores to avoid evicting it) + write x once
// = 256 MB ~ 11 us @ 23.3 TB/s. Iteration math is WMMA-bound and ~free.
// ---------------------------------------------------------------------------

typedef __bf16 v16bf __attribute__((ext_vector_type(16)));
typedef float  v8f   __attribute__((ext_vector_type(8)));
typedef float  v4f   __attribute__((ext_vector_type(4)));   // native vector for NT ld/st

#define NITER 20
#define EPS   1e-8f

__device__ __forceinline__ unsigned short f2bf(float x) {
  union { float f; unsigned u; } v; v.f = x;
  return (unsigned short)((v.u + 0x7fffu + ((v.u >> 16) & 1u)) >> 16);
}

// Assemble a 16-element bf16 WMMA fragment from two 16-byte LDS chunks.
__device__ __forceinline__ v16bf ldfrag(const unsigned short* p0,
                                        const unsigned short* p1) {
  union { v16bf v; uint4 q[2]; } t;
  t.q[0] = *(const uint4*)p0;
  t.q[1] = *(const uint4*)p1;
  return t.v;
}

struct __align__(16) Smem {
  unsigned short Ks[64 * 520];   // K slice, row-major, padded stride 520 halves
  unsigned short KT[512 * 72];   // K^T slice (512 rows x 64 cols), stride 72
  float cvec[512];               // c (f32 master copy)
  float vloc[512];               // local partial of v = K^T r
  float rloc[64];                // r for local rows
  float rowmax[64];              // row maxima (for exact exp in final pass)
  float upart[4][2][16];         // u partials: [i-tile][k-half][row]
  unsigned short cbf[512];       // c as bf16 (WMMA B operand)
  unsigned short rbf[64];        // r as bf16 (WMMA B operand)
};

__global__ __launch_bounds__(256)
void sinkhorn_kernel(const float* __restrict__ mat,
                     float* __restrict__ out,
                     unsigned* __restrict__ arrive,
                     float* __restrict__ vpart) {
  __shared__ Smem sm;

  const int t = threadIdx.x;
  const int w = t >> 5;          // wave 0..7
  const int l = t & 31;          // lane
  const int h = l >> 4;          // lane half
  const int m = l & 15;          // row-within-tile
  const int wg = blockIdx.x;
  const int b = wg >> 3;         // batch
  const int s = wg & 7;          // slice (64 rows)

  const float* matb = mat + ((size_t)(b * 512 + s * 64)) * 512;
  float*       outb = out + ((size_t)(b * 512 + s * 64)) * 512;

  // ------------------------------------------------------------------
  // Phase 1a: row pass. Each wave handles 8 rows; 32 lanes x 16 floats.
  // Computes rowmax, exp(z - max) -> bf16 K slice, row-sum -> r0.
  // (Regular-temporal loads: keep the matrix resident in L2 for phase 3.)
  // ------------------------------------------------------------------
  for (int rr = 0; rr < 8; ++rr) {
    const int r = w * 8 + rr;
    const float* rp = matb + (size_t)r * 512;

    float4 q[4];
#pragma unroll
    for (int qq = 0; qq < 4; ++qq)
      q[qq] = *(const float4*)(rp + qq * 128 + l * 4);

    float mx = -3.4e38f;
#pragma unroll
    for (int qq = 0; qq < 4; ++qq) {
      mx = fmaxf(mx, fmaxf(fmaxf(q[qq].x, q[qq].y), fmaxf(q[qq].z, q[qq].w)));
    }
#pragma unroll
    for (int k = 1; k < 32; k <<= 1) mx = fmaxf(mx, __shfl_xor(mx, k, 32));

    float sum = 0.f;
#pragma unroll
    for (int qq = 0; qq < 4; ++qq) {
      q[qq].x = __expf(q[qq].x - mx);
      q[qq].y = __expf(q[qq].y - mx);
      q[qq].z = __expf(q[qq].z - mx);
      q[qq].w = __expf(q[qq].w - mx);
      sum += q[qq].x + q[qq].y + q[qq].z + q[qq].w;
    }
#pragma unroll
    for (int k = 1; k < 32; k <<= 1) sum += __shfl_xor(sum, k, 32);

#pragma unroll
    for (int qq = 0; qq < 4; ++qq) {
      uint2 pk;
      pk.x = (unsigned)f2bf(q[qq].x) | ((unsigned)f2bf(q[qq].y) << 16);
      pk.y = (unsigned)f2bf(q[qq].z) | ((unsigned)f2bf(q[qq].w) << 16);
      *(uint2*)&sm.Ks[r * 520 + qq * 128 + l * 4] = pk;
    }
    if (l == 0) {
      sm.rowmax[r] = mx;
      sm.rloc[r]   = 512.0f / sum;    // x0 = softmax * m  ->  r0 = m / sumexp
    }
  }
  __syncthreads();

  // ------------------------------------------------------------------
  // Phase 1b: LDS transpose Ks -> KT, and init c = 1.
  // ------------------------------------------------------------------
  {
    const int i  = t & 63;
    const int jb = t >> 6;       // 0..3
    for (int j = jb * 128; j < jb * 128 + 128; j += 2) {
      unsigned pair = *(const unsigned*)&sm.Ks[i * 520 + j];
      sm.KT[(j)     * 72 + i] = (unsigned short)(pair & 0xffffu);
      sm.KT[(j + 1) * 72 + i] = (unsigned short)(pair >> 16);
    }
    for (int j = t; j < 512; j += 256) {
      sm.cvec[j] = 1.0f;
      sm.cbf[j]  = 0x3f80;       // bf16(1.0)
    }
  }

  // ------------------------------------------------------------------
  // Phase 2: 20 Sinkhorn iterations, WMMA matvecs.
  // ------------------------------------------------------------------
  for (int iter = 0; iter < NITER; ++iter) {
    __syncthreads();             // cbf / KT stable

    // ---- u = K * c  (wave w: i-tile w&3, k-half w>>2; 8 WMMAs each)
    {
      const int it = w & 3, kh = w >> 2;
      v8f acc = {};
      const unsigned short* rowp = sm.Ks + (it * 16 + m) * 520;
#pragma unroll
      for (int kk = 0; kk < 8; ++kk) {
        const int kt = kh * 8 + kk;
        v16bf A = ldfrag(rowp + kt * 32 + h * 8,
                         rowp + kt * 32 + h * 8 + 16);
        v16bf B = ldfrag(sm.cbf + kt * 32 + h * 8,
                         sm.cbf + kt * 32 + h * 8 + 16);
        acc = __builtin_amdgcn_wmma_f32_16x16x32_bf16(
            false, A, false, B, (short)0, acc, false, false);
      }
      if (l == 0) {
#pragma unroll
        for (int r2 = 0; r2 < 8; ++r2) sm.upart[it][kh][r2] = acc[r2];
      } else if (l == 16) {
#pragma unroll
        for (int r2 = 0; r2 < 8; ++r2) sm.upart[it][kh][8 + r2] = acc[r2];
      }
    }
    __syncthreads();

    // ---- r update (local 64 rows): r /= (r*u + eps)
    if (t < 64) {
      const float u  = sm.upart[t >> 4][0][t & 15] + sm.upart[t >> 4][1][t & 15];
      float ri = sm.rloc[t];
      ri = ri / (ri * u + EPS);
      sm.rloc[t] = ri;
      sm.rbf[t]  = f2bf(ri);
    }
    __syncthreads();

    // ---- v_partial = K^T(slice) * r  (wave w: j-tiles 4w..4w+3)
#pragma unroll
    for (int jj = 0; jj < 4; ++jj) {
      const int jt = w * 4 + jj;
      v8f acc = {};
      const unsigned short* rowp = sm.KT + (jt * 16 + m) * 72;
#pragma unroll
      for (int kt = 0; kt < 2; ++kt) {
        v16bf A = ldfrag(rowp + kt * 32 + h * 8,
                         rowp + kt * 32 + h * 8 + 16);
        v16bf B = ldfrag(sm.rbf + kt * 32 + h * 8,
                         sm.rbf + kt * 32 + h * 8 + 16);
        acc = __builtin_amdgcn_wmma_f32_16x16x32_bf16(
            false, A, false, B, (short)0, acc, false, false);
      }
      if (l == 0) {
#pragma unroll
        for (int r2 = 0; r2 < 8; ++r2) sm.vloc[jt * 16 + r2] = acc[r2];
      } else if (l == 16) {
#pragma unroll
        for (int r2 = 0; r2 < 8; ++r2) sm.vloc[jt * 16 + 8 + r2] = acc[r2];
      }
    }
    __syncthreads();

    // ---- publish partial v, 8-WG barrier (parity double buffer)
    const int par = iter & 1;
    float* gv = vpart + (((size_t)par * 128 + b) * 8 + s) * 512;
    if (t < 128) ((float4*)gv)[t] = ((const float4*)sm.vloc)[t];
    __threadfence();
    __syncthreads();
    if (t == 0) {
      atomicAdd(arrive + b, 1u);
      const unsigned target = 8u * (unsigned)(iter + 1);
      while (__hip_atomic_load(arrive + b, __ATOMIC_ACQUIRE,
                               __HIP_MEMORY_SCOPE_AGENT) < target)
        __builtin_amdgcn_s_sleep(2);
    }
    __syncthreads();

    // ---- c update (redundant, bit-identical in all 8 WGs of the batch)
    const float* gvb = vpart + (((size_t)par * 128 + b) * 8) * 512;
    for (int j = t; j < 512; j += 256) {
      float vs = 0.f;
#pragma unroll
      for (int sl = 0; sl < 8; ++sl) vs += gvb[sl * 512 + j];
      float cj = sm.cvec[j];
      cj = cj / (cj * vs + EPS);
      sm.cvec[j] = cj;
      sm.cbf[j]  = f2bf(cj);
    }
  }
  __syncthreads();

  // ------------------------------------------------------------------
  // Phase 3: out = r[i] * exp(z - rowmax[i]) * c[j]  (exact f32 kernel).
  // NT loads (matrix lines are dead after this) + NT stores (streaming
  // output must not evict other WGs' L2-resident matrix slices).
  // ------------------------------------------------------------------
  for (int rr = 0; rr < 8; ++rr) {
    const int r = w * 8 + rr;
    const float* rp = matb + (size_t)r * 512;
    float*       op = outb + (size_t)r * 512;
    const float mx = sm.rowmax[r];
    const float ri = sm.rloc[r];
#pragma unroll
    for (int qq = 0; qq < 4; ++qq) {
      const int col = qq * 128 + l * 4;
      v4f x4 = __builtin_nontemporal_load((const v4f*)(rp + col));
      v4f c4 = *(const v4f*)(sm.cvec + col);
      v4f o;
      o.x = __expf(x4.x - mx) * ri * c4.x;
      o.y = __expf(x4.y - mx) * ri * c4.y;
      o.z = __expf(x4.z - mx) * ri * c4.z;
      o.w = __expf(x4.w - mx) * ri * c4.w;
      __builtin_nontemporal_store(o, (v4f*)(op + col));
    }
  }
}

extern "C" void kernel_launch(void* const* d_in, const int* in_sizes, int n_in,
                              void* d_out, int out_size, void* d_ws, size_t ws_size,
                              hipStream_t stream) {
  (void)in_sizes; (void)n_in; (void)out_size; (void)ws_size;
  const float* mat = (const float*)d_in[0];
  float* out = (float*)d_out;

  // Workspace layout: [0,1024): per-batch arrival counters (128 u32, zeroed
  // each launch); [1024, 1024+4MB): parity-double-buffered v partials
  // [2][128 batches][8 slices][512] f32.
  unsigned* arrive = (unsigned*)d_ws;
  float* vpart = (float*)((char*)d_ws + 1024);
  (void)hipMemsetAsync(d_ws, 0, 1024, stream);

  sinkhorn_kernel<<<dim3(128 * 8), dim3(256), 0, stream>>>(mat, out, arrive, vpart);
}
